// CORALLoss_20315195310276
// MI455X (gfx1250) — compile-verified, matrix operand order checked by reading, not run on hardware
//
#include <hip/hip_runtime.h>
#include <math.h>

typedef float v2f  __attribute__((ext_vector_type(2)));
typedef float v8f  __attribute__((ext_vector_type(8)));
typedef _Float16 v16h __attribute__((ext_vector_type(16)));

#define NN   512
#define DD   128
#define TEMP_INV (1.0f/0.07f)

// cf[i] = features[i%256, i/256, :]  (features is (256,2,128) row-major)
__device__ __forceinline__ const float* cf_row(const float* f, int i) {
    return f + (size_t)(((i & 255) << 1) + (i >> 8)) * DD;
}

// ---------------- Kernel 1: per-row sqn, invnorm, proj --------------------
__global__ void coral_prep(const float* __restrict__ features,
                           const float* __restrict__ v_prog,
                           float* __restrict__ invnorm,
                           float* __restrict__ proj,
                           float* __restrict__ sqn) {
    __shared__ float r0[DD];
    __shared__ float r1[DD];
    const int i = blockIdx.x;
    const int t = threadIdx.x;           // 0..127
    const float x = cf_row(features, i)[t];
    const float v = v_prog[t];
    r0[t] = x * x;
    r1[t] = v * v;
    __syncthreads();
    for (int s = DD / 2; s > 0; s >>= 1) {
        if (t < s) { r0[t] += r0[t + s]; r1[t] += r1[t + s]; }
        __syncthreads();
    }
    const float s2 = r0[0];
    const float vv = r1[0];
    __syncthreads();
    r0[t] = x * v;
    __syncthreads();
    for (int s = DD / 2; s > 0; s >>= 1) {
        if (t < s) r0[t] += r0[t + s];
        __syncthreads();
    }
    if (t == 0) {
        sqn[i]     = s2;
        invnorm[i] = 1.0f / fmaxf(sqrtf(s2), 1e-12f);
        proj[i]    = r0[0] / (sqrtf(vv) + 1e-8f);   // cf . (v/(||v||+eps))
    }
}

// ---------------- Kernel 2: G = cf * cf^T via WMMA ------------------------
// One wave (32 lanes) per 16x16 output tile; 32x32 tiles; 8 waves/block.
__global__ void coral_gram_wmma(const float* __restrict__ features,
                                float* __restrict__ G) {
    const int wave = (blockIdx.x * blockDim.x + threadIdx.x) >> 5;  // 0..1023
    const int lane = threadIdx.x & 31;
    const int tr = wave >> 5;               // tile row 0..31
    const int tc = wave & 31;               // tile col 0..31
    const int rA = (tr << 4) + (lane & 15);
    const int rB = (tc << 4) + (lane & 15);
    const float* pa = cf_row(features, rA);
    const float* pb = cf_row(features, rB);
    v8f acc = {};

#if __has_builtin(__builtin_amdgcn_wmma_f32_16x16x4_f32)
    // fp32 A(16x4): lanes 0-15 -> K = k0+{0,1}; lanes 16-31 -> K = k0+{2,3}
    const int koff = (lane >> 4) << 1;
    const float* qa = pa + koff;
    const float* qb = pb + koff;
    #pragma unroll
    for (int s = 0; s < DD / 4; ++s) {
        v2f a = *(const v2f*)(qa + 4 * s);
        v2f b = *(const v2f*)(qb + 4 * s);
        acc = __builtin_amdgcn_wmma_f32_16x16x4_f32(
            false, a, false, b, (short)0, acc, false, false);
    }
#else
    // Fallback: codegen-confirmed f16 WMMA (16x16x32), convert fp32 -> fp16.
    // 16-bit A(16x32) layout: lanes 0-15: K = kb + {0..7, 16..23};
    //                          lanes 16-31: K = kb + {8..15, 24..31}.
    const int kbase = (lane >> 4) << 3;
    #pragma unroll
    for (int kb = 0; kb < DD; kb += 32) {
        v16h a, b;
        #pragma unroll
        for (int e = 0; e < 8; ++e) {
            a[e]     = (_Float16)pa[kb + kbase + e];
            a[e + 8] = (_Float16)pa[kb + kbase + 16 + e];
            b[e]     = (_Float16)pb[kb + kbase + e];
            b[e + 8] = (_Float16)pb[kb + kbase + 16 + e];
        }
        acc = __builtin_amdgcn_wmma_f32_16x16x32_f16(
            false, a, false, b, (short)0, acc, false, false);
    }
#endif

    // C/D f32 16x16: VGPR v: lanes 0-15 -> (M=v, N=lane); lanes 16-31 -> (M=8+v, N=lane-16)
    const int Mbase = (tr << 4) + ((lane >> 4) << 3);
    const int col   = (tc << 4) + (lane & 15);
    #pragma unroll
    for (int v = 0; v < 8; ++v)
        G[(size_t)(Mbase + v) * NN + col] = acc[v];
}

// ---------------- Kernel 3: per-row RAL + OAL partials --------------------
__global__ void coral_rowloss(const float* __restrict__ labels,
                              const float* __restrict__ G,
                              const float* __restrict__ invnorm,
                              const float* __restrict__ proj,
                              const float* __restrict__ sqn,
                              float* __restrict__ row_out) {  // 4*NN floats
    __shared__ float sim_row[NN];
    __shared__ float exp_row[NN];
    __shared__ float pd_row[NN];
    __shared__ float pos_row[NN];
    __shared__ float4 red[256];

    const int i = blockIdx.x;
    const int t = threadIdx.x;             // 0..255
    const float pos_i = labels[i & 255];
    const float inv_i = invnorm[i];

    for (int k = t; k < NN; k += 256) {
        const float g  = G[(size_t)i * NN + k];
        const float s  = g * inv_i * invnorm[k] * TEMP_INV;
        const float pk = labels[k & 255];
        sim_row[k] = s;
        exp_row[k] = expf(s);
        pd_row[k]  = fabsf(pos_i - pk);
        pos_row[k] = pk;
    }
    __syncthreads();

    const float sqn_i  = sqn[i];
    const float proj_i = proj[i];
    float ral_s = 0.0f, ral_c = 0.0f, oal_s = 0.0f, oal_c = 0.0f;

    for (int j = t; j < NN; j += 256) {
        const float pdij = pd_row[j];
        if (j != i) {
            // denom[i,j] = sum_{k!=i, pd[i,k] >= pd[i,j]} exp(sim[i,k])
            float denom = 0.0f;
            for (int k = 0; k < NN; ++k) {
                if (k != i && pd_row[k] >= pdij) denom += exp_row[k];
            }
            if (denom > 0.0f) {
                const float sw = 1.0f / (1.0f + expf(-pdij));   // sigmoid(pd)
                ral_s += (logf(denom + 1e-8f) - sim_row[j]) * sw;
                ral_c += 1.0f;
            }
        }
        if (pos_i < pos_row[j]) {
            const float g    = G[(size_t)i * NN + j];
            const float sqd  = fmaxf(sqn_i + sqn[j] - 2.0f * g, 0.0f);
            const float dist = sqrtf(fmaxf(sqd, 1e-24f));
            oal_s += (proj[j] - proj_i) / fmaxf(dist, 1e-12f);
            oal_c += 1.0f;
        }
    }

    red[t] = make_float4(ral_s, ral_c, oal_s, oal_c);
    __syncthreads();
    for (int s = 128; s > 0; s >>= 1) {
        if (t < s) {
            red[t].x += red[t + s].x; red[t].y += red[t + s].y;
            red[t].z += red[t + s].z; red[t].w += red[t + s].w;
        }
        __syncthreads();
    }
    if (t == 0) {
        row_out[i]          = red[0].x;
        row_out[NN + i]     = red[0].y;
        row_out[2 * NN + i] = red[0].z;
        row_out[3 * NN + i] = red[0].w;
    }
}

// ---------------- Kernel 4: final scalar ----------------------------------
__global__ void coral_final(const float* __restrict__ row_out,
                            float* __restrict__ out) {
    __shared__ float4 red[NN];
    const int t = threadIdx.x;             // 0..511
    red[t] = make_float4(row_out[t], row_out[NN + t],
                         row_out[2 * NN + t], row_out[3 * NN + t]);
    __syncthreads();
    for (int s = NN / 2; s > 0; s >>= 1) {
        if (t < s) {
            red[t].x += red[t + s].x; red[t].y += red[t + s].y;
            red[t].z += red[t + s].z; red[t].w += red[t + s].w;
        }
        __syncthreads();
    }
    if (t == 0) {
        const float ral = (red[0].y > 0.0f) ? red[0].x / red[0].y : 0.0f;
        const float oal = (red[0].w > 0.0f) ? -red[0].z / red[0].w : 0.0f;
        out[0] = ral + oal;   // ALPHA = 1.0
    }
}

// ---------------- Host launcher -------------------------------------------
extern "C" void kernel_launch(void* const* d_in, const int* in_sizes, int n_in,
                              void* d_out, int out_size, void* d_ws, size_t ws_size,
                              hipStream_t stream) {
    const float* features = (const float*)d_in[0];  // (256,2,128)
    const float* labels   = (const float*)d_in[1];  // (256,)
    const float* v_prog   = (const float*)d_in[2];  // (128,)
    float* out = (float*)d_out;

    float* ws      = (float*)d_ws;
    float* G       = ws;                       // NN*NN
    float* invnorm = G + (size_t)NN * NN;      // NN
    float* proj    = invnorm + NN;             // NN
    float* sqn     = proj + NN;                // NN
    float* row_out = sqn + NN;                 // 4*NN

    coral_prep<<<NN, DD, 0, stream>>>(features, v_prog, invnorm, proj, sqn);
    coral_gram_wmma<<<(NN / 16) * (NN / 16) / 8, 256, 0, stream>>>(features, G);
    coral_rowloss<<<NN, 256, 0, stream>>>(labels, G, invnorm, proj, sqn, row_out);
    coral_final<<<1, NN, 0, stream>>>(row_out, out);
    (void)in_sizes; (void)n_in; (void)out_size; (void)ws_size;
}
